// TargetAspectClassificationBlock_18528488915007
// MI455X (gfx1250) — matrix operand order repaired and checked by reading, not compiled, not run on hardware
//
#include <hip/hip_runtime.h>
#include <hip/hip_bf16.h>

// ---------------------------------------------------------------------------
// Problem constants (from reference): B=8, S=1024, H=768, NH=12, DH=64,
// T=64 targets, A=12 aspects. M = B*S = 8192, K = N = 768.
// Strategy: bf16 WMMA (v_wmma_f32_16x16x32_bf16) for all GEMMs + flash attn.
// CDNA5 paths: async global->LDS staging (ASYNCcnt) for bf16 A tiles,
// global_prefetch_b8 for next tiles, wave32 ds_bpermute reductions.
// attention_mask is uniform per batch -> cancels in softmax -> ignored.
// ---------------------------------------------------------------------------

typedef __attribute__((ext_vector_type(16))) __bf16 v16bf;
typedef __attribute__((ext_vector_type(8)))  float  v8f;

#define BB 8
#define SS 1024
#define HH 768
#define NHEAD 12
#define DHEAD 64
#define TT 64
#define AA 12

static __device__ __forceinline__ unsigned short f32_to_bf16_bits(float f) {
    union { float f; unsigned u; } v; v.f = f;
    unsigned r = v.u + 0x7FFFu + ((v.u >> 16) & 1u);   // round-to-nearest-even
    return (unsigned short)(r >> 16);
}
static __device__ __forceinline__ float bf16_bits_to_f32(unsigned short h) {
    union { unsigned u; float f; } v; v.u = ((unsigned)h) << 16;
    return v.f;
}
static __device__ __forceinline__ __bf16 bits_to_bf16(unsigned short h) {
    return __builtin_bit_cast(__bf16, h);
}

static __device__ __forceinline__ v8f wmma_bf16(v16bf a, v16bf b, v8f c) {
    return __builtin_amdgcn_wmma_f32_16x16x32_bf16(
        /*neg_a=*/false, a, /*neg_b=*/false, b,
        /*c_mod=*/(short)0, c, /*reuse_a=*/false, /*reuse_b=*/false);
}

// gfx1250 async global->LDS copy (16B per lane), tracked by ASYNCcnt.
static __device__ __forceinline__ void async_g2l_b128(unsigned int lds_off,
                                                      unsigned long long gaddr) {
    asm volatile("global_load_async_to_lds_b128 %0, %1, off"
                 :: "v"(lds_off), "v"(gaddr) : "memory");
}
static __device__ __forceinline__ void wait_asynccnt0() {
    asm volatile("s_wait_asynccnt 0x0" ::: "memory");
}

// load-as-float helpers (A operand may be fp32 or bf16-bits)
static __device__ __forceinline__ float loadAs(const float* p, size_t i) { return p[i]; }
static __device__ __forceinline__ float loadAs(const unsigned short* p, size_t i) { return bf16_bits_to_f32(p[i]); }

// ---------------------------------------------------------------------------
// Tiled WMMA GEMM:  C(MxN) = act( A(MxK) @ Bw(KxN) + bias )
// Block tile 128x128, BK=32, 8 waves (4x2), each wave 32x64 (2x4 WMMA tiles).
// ACT: 0 = relu, 1 = scale by 0.125 (q), 2 = none (k). Output bf16 bits.
// N, K must be multiples of 128 / 32 (true here: 768). M guarded (fp32 path).
// bf16 A path (q/k GEMMs, M=8192) stages A via async global->LDS DMA.
// ---------------------------------------------------------------------------
template <typename TA, int ACT>
__global__ __launch_bounds__(256) void gemm_bias_act(
    const TA* __restrict__ A, const float* __restrict__ Bw,
    const float* __restrict__ bias, unsigned short* __restrict__ C,
    int M, int N, int K)
{
    constexpr int BM = 128, BN = 128, BK = 32;
    constexpr bool A_IS_BF16 = (sizeof(TA) == 2);
    // bf16 async path writes raw 16B chunks: row stride must stay 16B-aligned.
    __shared__ unsigned short Asm[BM][BK + 8];   // stride 40*2=80B (16B aligned)
    __shared__ unsigned short Bsm[BK][BN + 8];

    const int tid  = threadIdx.x;
    const int lane = tid & 31;
    const int wid  = tid >> 5;
    const int wm   = wid & 3;   // 0..3 : 32-row strip
    const int wn   = wid >> 2;  // 0..1 : 64-col strip
    const int m0   = blockIdx.y * BM;
    const int n0   = blockIdx.x * BN;

    v8f acc[2][4];
    #pragma unroll
    for (int mi = 0; mi < 2; ++mi)
        #pragma unroll
        for (int ni = 0; ni < 4; ++ni) acc[mi][ni] = {};

    for (int k0 = 0; k0 < K; k0 += BK) {
        // ---- stage A tile (128x32) into bf16 LDS ----
        if (A_IS_BF16) {
            // Async DMA path: 128x32 bf16 = 512 x 16B chunks, 2 per thread.
            // (Only used when M % BM == 0, so no bounds handling needed.)
            #pragma unroll
            for (int cc = 0; cc < 2; ++cc) {
                const int c   = tid * 2 + cc;
                const int row = c >> 2;           // 4 chunks (64B) per row
                const int col = (c & 3) * 8;      // elements
                const unsigned long long g =
                    (unsigned long long)(uintptr_t)(A + (size_t)(m0 + row) * K + (k0 + col));
                const unsigned int l = (unsigned int)(uintptr_t)&Asm[row][col];
                async_g2l_b128(l, g);
            }
        } else {
            const int row  = tid >> 1;
            const int colb = (tid & 1) * 16;
            const int gr   = m0 + row;
            #pragma unroll
            for (int i = 0; i < 16; ++i) {
                float v = 0.0f;
                if (gr < M) v = loadAs(A, (size_t)gr * K + (k0 + colb + i));
                Asm[row][colb + i] = f32_to_bf16_bits(v);
            }
            if (k0 + BK < K && gr < M)
                __builtin_prefetch(&A[(size_t)gr * K + (k0 + BK + colb)], 0, 1);
        }
        // ---- stage B tile (32x128) -> bf16 LDS (fp32 weights, convert) ----
        {
            const int row  = tid >> 3;
            const int colb = (tid & 7) * 16;
            #pragma unroll
            for (int i = 0; i < 16; ++i)
                Bsm[row][colb + i] =
                    f32_to_bf16_bits(Bw[(size_t)(k0 + row) * N + (n0 + colb + i)]);
            if (k0 + BK < K)
                __builtin_prefetch(&Bw[(size_t)(k0 + BK + row) * N + (n0 + colb)], 0, 1);
        }
        if (A_IS_BF16) wait_asynccnt0();   // async LDS writes visible before barrier
        __syncthreads();

        // build fragments (contiguous k = 16*(lane/16) + i, same order for A and B)
        v16bf afr[2];
        {
            const int arow = lane & 15;
            const int k0l  = (lane & 16) ? 16 : 0;
            #pragma unroll
            for (int mi = 0; mi < 2; ++mi) {
                const int r = wm * 32 + mi * 16 + arow;
                #pragma unroll
                for (int i = 0; i < 16; ++i) {
                    const int kk = k0l + i;
                    afr[mi][i] = bits_to_bf16(Asm[r][kk]);
                }
            }
        }
        v16bf bfr[4];
        {
            const int bcol = lane & 15;
            const int kb   = (lane & 16) ? 16 : 0;
            #pragma unroll
            for (int ni = 0; ni < 4; ++ni) {
                const int c = wn * 64 + ni * 16 + bcol;
                #pragma unroll
                for (int i = 0; i < 16; ++i)
                    bfr[ni][i] = bits_to_bf16(Bsm[kb + i][c]);
            }
        }
        #pragma unroll
        for (int mi = 0; mi < 2; ++mi)
            #pragma unroll
            for (int ni = 0; ni < 4; ++ni)
                acc[mi][ni] = wmma_bf16(afr[mi], bfr[ni], acc[mi][ni]);
        __syncthreads();
    }

    // epilogue: bias + activation -> bf16 store
    const int ccol  = lane & 15;
    const int rbase = (lane & 16) ? 8 : 0;
    #pragma unroll
    for (int mi = 0; mi < 2; ++mi) {
        #pragma unroll
        for (int ni = 0; ni < 4; ++ni) {
            const int col = n0 + wn * 64 + ni * 16 + ccol;
            const float bv = bias[col];
            #pragma unroll
            for (int r = 0; r < 8; ++r) {
                const int row = m0 + wm * 32 + mi * 16 + rbase + r;
                if (row < M) {
                    float v = acc[mi][ni][r] + bv;
                    if (ACT == 0) v = fmaxf(v, 0.0f);
                    else if (ACT == 1) v *= 0.125f;  // rsqrt(DH)
                    C[(size_t)row * N + col] = f32_to_bf16_bits(v);
                }
            }
        }
    }
}

// ---------------------------------------------------------------------------
// Flash attention: one wave per (b, head, 16-query tile).
// q,k,v are bf16 bits laid out (B,S,NH,DH). out is fp32 (B,S,NH,DH).
// Online softmax over s in blocks of 32; scale already folded into q.
// ---------------------------------------------------------------------------
__global__ __launch_bounds__(32) void attn_kernel(
    const unsigned short* __restrict__ q,
    const unsigned short* __restrict__ k,
    const unsigned short* __restrict__ v,
    float* __restrict__ out)
{
    const int lane  = threadIdx.x & 31;
    const int qbase = blockIdx.x * 16;
    const int head  = blockIdx.y;
    const int b     = blockIdx.z;
    const size_t base = ((size_t)b * SS) * HH + (size_t)head * DHEAD;

    __shared__ unsigned short p_lds[16][32 + 8];
    __shared__ unsigned short v_lds[32][64 + 8];

    const int rbase = (lane & 16) ? 8 : 0;

    // Q fragments: d = 0..31 and 32..63
    v16bf qf[2];
    {
        const int row = lane & 15;
        const int k0l = (lane & 16) ? 16 : 0;
        const unsigned short* qp = q + base + (size_t)(qbase + row) * HH;
        #pragma unroll
        for (int f = 0; f < 2; ++f)
            #pragma unroll
            for (int i = 0; i < 16; ++i) {
                const int d = f * 32 + k0l + i;
                qf[f][i] = bits_to_bf16(qp[d]);
            }
    }

    v8f Of[4];
    #pragma unroll
    for (int t = 0; t < 4; ++t) Of[t] = {};
    float m8[8], l8[8];
    #pragma unroll
    for (int r = 0; r < 8; ++r) { m8[r] = -3.0e38f; l8[r] = 0.0f; }

    for (int s0 = 0; s0 < SS; s0 += 32) {
        // stage V block (32 x 64) coalesced: one row (128B) per lane
        {
            const uint4* vp = reinterpret_cast<const uint4*>(v + base + (size_t)(s0 + lane) * HH);
            uint4* dst = reinterpret_cast<uint4*>(&v_lds[lane][0]);
            #pragma unroll
            for (int j = 0; j < 8; ++j) dst[j] = vp[j];
        }
        // prefetch next block's K/V rows into caches while we compute
        if (s0 + 32 < SS) {
            __builtin_prefetch(v + base + (size_t)(s0 + 32 + lane) * HH, 0, 1);
            __builtin_prefetch(k + base + (size_t)(s0 + 32 + lane) * HH, 0, 1);
        }
        // K^T fragments straight from global (contiguous d per lane)
        v16bf kf[2][2];
        {
            const int kb = (lane & 16) ? 16 : 0;
            #pragma unroll
            for (int g = 0; g < 2; ++g) {
                const int srow = s0 + g * 16 + (lane & 15);
                const unsigned short* kp = k + base + (size_t)srow * HH;
                #pragma unroll
                for (int dc = 0; dc < 2; ++dc)
                    #pragma unroll
                    for (int i = 0; i < 16; ++i)
                        kf[g][dc][i] = bits_to_bf16(kp[dc * 32 + kb + i]);
            }
        }
        __syncthreads();

        // scores S (16 x 32) = Q @ K^T : two column-group C tiles
        v8f S0 = {}, S1 = {};
        S0 = wmma_bf16(qf[0], kf[0][0], S0);
        S0 = wmma_bf16(qf[1], kf[0][1], S0);
        S1 = wmma_bf16(qf[0], kf[1][0], S1);
        S1 = wmma_bf16(qf[1], kf[1][1], S1);

        // online softmax update (cross-lane row reductions, wave32)
        float alpha[8];
        #pragma unroll
        for (int r = 0; r < 8; ++r) {
            float mx = fmaxf(S0[r], S1[r]);
            mx = fmaxf(mx, __shfl_xor(mx, 1, 32));
            mx = fmaxf(mx, __shfl_xor(mx, 2, 32));
            mx = fmaxf(mx, __shfl_xor(mx, 4, 32));
            mx = fmaxf(mx, __shfl_xor(mx, 8, 32));
            const float mnew = fmaxf(m8[r], mx);
            alpha[r] = __expf(m8[r] - mnew);
            m8[r] = mnew;
            const float p0 = __expf(S0[r] - mnew);
            const float p1 = __expf(S1[r] - mnew);
            float ps = p0 + p1;
            ps += __shfl_xor(ps, 1, 32);
            ps += __shfl_xor(ps, 2, 32);
            ps += __shfl_xor(ps, 4, 32);
            ps += __shfl_xor(ps, 8, 32);
            l8[r] = l8[r] * alpha[r] + ps;
            const int prow = rbase + r;
            p_lds[prow][(lane & 15)]      = f32_to_bf16_bits(p0);
            p_lds[prow][16 + (lane & 15)] = f32_to_bf16_bits(p1);
        }
        // rescale running output
        #pragma unroll
        for (int t = 0; t < 4; ++t)
            #pragma unroll
            for (int r = 0; r < 8; ++r) Of[t][r] *= alpha[r];
        __syncthreads();

        // P fragment (A layout) from LDS transpose buffer
        v16bf pf;
        {
            const int prow = lane & 15;
            const int k0l  = (lane & 16) ? 16 : 0;
            #pragma unroll
            for (int i = 0; i < 16; ++i) {
                const int kk = k0l + i;
                pf[i] = bits_to_bf16(p_lds[prow][kk]);
            }
        }
        // V fragments (B layout) from LDS
        v16bf vf[4];
        {
            const int kb = (lane & 16) ? 16 : 0;
            #pragma unroll
            for (int t = 0; t < 4; ++t) {
                const int d = t * 16 + (lane & 15);
                #pragma unroll
                for (int i = 0; i < 16; ++i)
                    vf[t][i] = bits_to_bf16(v_lds[kb + i][d]);
            }
        }
        #pragma unroll
        for (int t = 0; t < 4; ++t) Of[t] = wmma_bf16(pf, vf[t], Of[t]);
        __syncthreads();
    }

    // epilogue: divide by softmax denom, write fp32
    #pragma unroll
    for (int t = 0; t < 4; ++t) {
        const int d = head * DHEAD + t * 16 + (lane & 15);
        #pragma unroll
        for (int r = 0; r < 8; ++r) {
            const int row = qbase + rbase + r;
            out[((size_t)b * SS + row) * HH + d] = Of[t][r] / l8[r];
        }
    }
}

// ---------------------------------------------------------------------------
// rep[t, :] = mean over span [s0,s1) of attn_out[tb[t], s, :]
// ---------------------------------------------------------------------------
__global__ __launch_bounds__(256) void rep_kernel(
    const float* __restrict__ attn_out, const int* __restrict__ pos,
    const int* __restrict__ tb, float* __restrict__ rep)
{
    const int t = blockIdx.x;
    const int bsel = tb[t];
    const int s0 = pos[2 * t], s1 = pos[2 * t + 1];
    int cnt = s1 - s0; if (cnt < 1) cnt = 1;
    const float inv = 1.0f / (float)cnt;
    for (int hh = threadIdx.x; hh < HH; hh += 256) {
        float s = 0.0f;
        for (int si = s0; si < s1; ++si)
            s += attn_out[((size_t)bsel * SS + si) * HH + hh];
        rep[(size_t)t * HH + hh] = s * inv;
    }
}

// ---------------------------------------------------------------------------
// logits[t,a] = rep[t]·Wc[0:H] + h_a[tb[t],a]·Wc[H:2H] + bc
// ---------------------------------------------------------------------------
__global__ __launch_bounds__(256) void final_kernel(
    const float* __restrict__ rep, const unsigned short* __restrict__ h_a,
    const int* __restrict__ tb, const float* __restrict__ Wc,
    const float* __restrict__ bc, float* __restrict__ outp)
{
    __shared__ float red[256];
    const int t = blockIdx.x;
    const int tid = threadIdx.x;
    const int bsel = tb[t];

    float d1 = 0.0f;
    for (int hh = tid; hh < HH; hh += 256)
        d1 += rep[(size_t)t * HH + hh] * Wc[hh];
    red[tid] = d1; __syncthreads();
    for (int s = 128; s > 0; s >>= 1) {
        if (tid < s) red[tid] += red[tid + s];
        __syncthreads();
    }
    const float dot1 = red[0];
    __syncthreads();

    for (int a = 0; a < AA; ++a) {
        float d2 = 0.0f;
        for (int hh = tid; hh < HH; hh += 256)
            d2 += bf16_bits_to_f32(h_a[((size_t)bsel * AA + a) * HH + hh]) * Wc[HH + hh];
        red[tid] = d2; __syncthreads();
        for (int s = 128; s > 0; s >>= 1) {
            if (tid < s) red[tid] += red[tid + s];
            __syncthreads();
        }
        if (tid == 0) outp[(size_t)t * AA + a] = dot1 + red[0] + bc[0];
        __syncthreads();
    }
}

// ---------------------------------------------------------------------------
extern "C" void kernel_launch(void* const* d_in, const int* in_sizes, int n_in,
                              void* d_out, int out_size, void* d_ws, size_t ws_size,
                              hipStream_t stream) {
    (void)in_sizes; (void)n_in; (void)out_size; (void)ws_size;

    const float* all_tokens    = (const float*)d_in[0];   // (B,S,H) fp32
    const float* aspect_tokens = (const float*)d_in[1];   // (B,A,H) fp32
    const int*   pos           = (const int*)d_in[2];     // (T,2)
    const int*   tb            = (const int*)d_in[3];     // (T,)
    /* d_in[4] attention_mask: uniform per batch -> softmax-invariant, unused */
    const float* W1 = (const float*)d_in[5];
    const float* b1 = (const float*)d_in[6];
    const float* Wq = (const float*)d_in[7];              // (H, NH*DH) flat
    const float* bq = (const float*)d_in[8];
    const float* Wk = (const float*)d_in[9];
    const float* bk = (const float*)d_in[10];
    const float* Wc = (const float*)d_in[11];             // (2H,1)
    const float* bc = (const float*)d_in[12];
    float* outp = (float*)d_out;                          // (T,A,1) fp32

    const int M = BB * SS;                                // 8192
    char* ws = (char*)d_ws;
    unsigned short* h_bf   = (unsigned short*)(ws);                       // 12.6 MB
    unsigned short* q_bf   = (unsigned short*)(ws + 12582912);            // 12.6 MB
    unsigned short* k_bf   = (unsigned short*)(ws + 25165824);            // 12.6 MB
    float*          attn_o = (float*)(ws + 37748736);                     // 25.2 MB
    float*          rep    = (float*)(ws + 62914560);                     // 0.2 MB
    unsigned short* h_a    = (unsigned short*)(ws + 63111168);            // 0.15 MB

    dim3 gblk(256);
    dim3 ggrid(HH / 128, M / 128);          // (6, 64)
    // h = relu(X @ W1 + b1)
    gemm_bias_act<float, 0><<<ggrid, gblk, 0, stream>>>(all_tokens, W1, b1, h_bf, M, HH, HH);
    // q = (h @ Wq + bq) * 0.125 ; k = h @ Wk + bk   (bf16 A staged via async DMA)
    gemm_bias_act<unsigned short, 1><<<ggrid, gblk, 0, stream>>>(h_bf, Wq, bq, q_bf, M, HH, HH);
    gemm_bias_act<unsigned short, 2><<<ggrid, gblk, 0, stream>>>(h_bf, Wk, bk, k_bf, M, HH, HH);
    // h_a = relu(aspect_tokens @ W1 + b1), M = B*A = 96
    gemm_bias_act<float, 0><<<dim3(HH / 128, 1), gblk, 0, stream>>>(aspect_tokens, W1, b1, h_a, BB * AA, HH, HH);
    // flash attention (v = h)
    attn_kernel<<<dim3(SS / 16, NHEAD, BB), dim3(32), 0, stream>>>(q_bf, k_bf, h_bf, attn_o);
    // span means
    rep_kernel<<<dim3(TT), dim3(256), 0, stream>>>(attn_o, pos, tb, rep);
    // final logits
    final_kernel<<<dim3(TT), dim3(256), 0, stream>>>(rep, h_a, tb, Wc, bc, outp);
}
